// DVBF_57956288692444
// MI455X (gfx1250) — matrix operand order repaired and checked by reading, not compiled, not run on hardware
//
#include <hip/hip_runtime.h>

// ---------------------------------------------------------------------------
// DVBF for MI455X (gfx1250, wave32). All matmuls run on v_wmma_f32_16x16x32_f16
// (f16 multiply inputs, f32 accumulation). Recurrences are batch-independent,
// so each block scans all T steps for a 16-row batch tile with state in LDS
// (dynamic LDS, using the 320KB/WGP budget). No atomics: deterministic
// fixed-order reductions only. Workspace use ~171 MB (xg buffer reused as r1).
// ---------------------------------------------------------------------------

typedef __attribute__((ext_vector_type(16))) _Float16 v16h;
typedef __attribute__((ext_vector_type(8)))  float    v8f;

#define N_  512
#define T_  128
#define DX_ 256
#define DU_ 32
#define DZ_ 128
#define DW_ 64
#define M_  16
#define H_  128

__device__ __forceinline__ float sigmoidf_(float x) { return 1.0f / (1.0f + expf(-x)); }

__device__ __forceinline__ v8f wmma_(v16h a, v16h b, v8f c) {
  return __builtin_amdgcn_wmma_f32_16x16x32_f16(false, a, false, b, (short)0, c, false, false);
}

// A fragment (16x32, f16) from row-major fp32 source. src = &Mat[tile_row0][k0].
// Contiguous K order (same as B): lanes0-15 row M=lane, K in {0..15}; lanes16-31 K+=16.
__device__ __forceinline__ v16h load_fragA_f32(const float* src, int ld) {
  const int l = threadIdx.x & 31;
  const float* p = src + (l & 15) * ld + ((l >> 4) << 4);
  v16h f;
#pragma unroll
  for (int i = 0; i < 16; ++i) f[i] = (_Float16)p[i];
  return f;
}

// B fragment (32x16, f16) for X @ W^T: B column n = W row. src = &W[col0][k0].
// Contiguous K order (same as A): lane holds column n = lane&15, 16 contiguous K (lanes>=16: K+=16).
__device__ __forceinline__ v16h load_fragB_f32(const float* src, int ld) {
  const int l = threadIdx.x & 31;
  const float* p = src + (l & 15) * ld + ((l >> 4) << 4);
  v16h f;
#pragma unroll
  for (int i = 0; i < 16; ++i) f[i] = (_Float16)p[i];
  return f;
}

__device__ __forceinline__ v16h load_fragB_f16(const _Float16* src, int ld) {
  const int l = threadIdx.x & 31;
  const _Float16* p = src + (l & 15) * ld + ((l >> 4) << 4);
  v16h f;
#pragma unroll
  for (int i = 0; i < 16; ++i) f[i] = p[i];
  return f;
}

// C/D tile: VGPR v -> row v + 8*(lane>=16), col = lane&15.
__device__ __forceinline__ v8f load_ctile(const float* src, int ld) {
  const int l = threadIdx.x & 31;
  const int col = l & 15, rb = ((l >> 4) & 1) << 3;
  v8f c;
#pragma unroll
  for (int v = 0; v < 8; ++v) c[v] = src[(rb + v) * ld + col];
  return c;
}

__device__ __forceinline__ void store_tile(float* dst, int ld, v8f acc) {
  const int l = threadIdx.x & 31;
  const int col = l & 15, rb = ((l >> 4) & 1) << 3;
#pragma unroll
  for (int v = 0; v < 8; ++v) dst[(rb + v) * ld + col] = acc[v];
}

// B fragment for the stacked [A | B | C] mixture weight (row c = m*128+i,
// K concat: z(0..127) | u(128..159) | w(160..223)). Each 32-chunk lies in one
// segment, so the segment choice is uniform per fragment element group.
__device__ __forceinline__ v16h load_fragB_abc(const float* Am, const float* Bm,
                                               const float* Cm, int c0, int k0) {
  const int l = threadIdx.x & 31;
  const int r = c0 + (l & 15);
  const int mm = r >> 7, ii = r & 127;
  const int kb = k0 + ((l >> 4) << 4);
  const float* p;
  int kloc;
  if (kb < 128)      { p = Am + ((long)mm * 128 + ii) * 128; kloc = kb; }
  else if (kb < 160) { p = Bm + ((long)mm * 128 + ii) * 32;  kloc = kb - 128; }
  else               { p = Cm + ((long)mm * 128 + ii) * 64;  kloc = kb - 160; }
  v16h f;
#pragma unroll
  for (int i = 0; i < 16; ++i) f[i] = (_Float16)p[kloc + i];
  return f;
}

// ------------------------- generic wave-tile GEMM --------------------------
// C[rows,cols] = act(A[rows,K] @ W[cols,K]^T + bias). One 16x16 tile per wave,
// grid-stride. act: 0 = none, 1 = relu.
__global__ __launch_bounds__(256)
void k_gemm(const float* __restrict__ A, long lda,
            const float* __restrict__ W, long ldw,
            float* __restrict__ C, long ldc,
            const float* __restrict__ bias,
            int rows, int cols, int K, int act) {
  const int gw = (int)((blockIdx.x * blockDim.x + threadIdx.x) >> 5);
  const int nw = (int)((gridDim.x * blockDim.x) >> 5);
  const int ct = cols >> 4;
  const long total = (long)(rows >> 4) * ct;
  const int lane = threadIdx.x & 31;
  const int rb = ((lane >> 4) & 1) << 3;
  for (long ti = gw; ti < total; ti += nw) {
    const int r0 = (int)(ti / ct) << 4;
    const int c0 = (int)(ti % ct) << 4;
    v8f acc = {};
    for (int k0 = 0; k0 < K; k0 += 32) {
      v16h a = load_fragA_f32(A + (long)r0 * lda + k0, (int)lda);
      v16h b = load_fragB_f32(W + (long)c0 * ldw + k0, (int)ldw);
      acc = wmma_(a, b, acc);
    }
    const int col = lane & 15;
    const float bv = bias ? bias[c0 + col] : 0.0f;
    float* dst = C + (long)r0 * ldc + c0;
#pragma unroll
    for (int v = 0; v < 8; ++v) {
      float o = acc[v] + bv;
      if (act == 1) o = fmaxf(o, 0.0f);
      dst[(long)(rb + v) * ldc + col] = o;
    }
  }
}

// -------------------------- forward LSTM scan ------------------------------
// Block = 16 batch rows, loops t=0..T-1. xg = x@Wih^T + b precomputed.
// Whh cached in LDS as f16 (128KB); gates accumulate onto xg tile via WMMA C.
__global__ __launch_bounds__(256)
void k_lstm_fw(const float* __restrict__ xg, const float* __restrict__ Whh,
               float* __restrict__ h0) {
  extern __shared__ float smem[];
  float* hbuf  = smem;               // 16*128
  float* cbuf  = hbuf + 2048;        // 16*128
  float* gates = cbuf + 2048;        // 16*512
  _Float16* wh = (_Float16*)(gates + 8192);  // 512*128 f16
  const int tid = threadIdx.x;
  const int wv = tid >> 5;
  const int n0 = blockIdx.x * 16;
  for (int i = tid; i < 512 * 128; i += 256) wh[i] = (_Float16)Whh[i];
  for (int i = tid; i < 2048; i += 256) { hbuf[i] = 0.0f; cbuf[i] = 0.0f; }
  __syncthreads();
  const int lane = tid & 31;
  const int rb = ((lane >> 4) & 1) << 3;
  for (int t = 0; t < T_; ++t) {
    v16h af[4];
#pragma unroll
    for (int kc = 0; kc < 4; ++kc) af[kc] = load_fragA_f32(hbuf + kc * 32, 128);
    for (int q = 0; q < 4; ++q) {
      const int c0 = ((wv << 2) + q) << 4;
      const float* xsrc = xg + ((long)n0 * T_ + t) * 512 + c0;  // row stride T*512
      v8f acc = load_ctile(xsrc, T_ * 512);
#pragma unroll
      for (int kc = 0; kc < 4; ++kc)
        acc = wmma_(af[kc], load_fragB_f16(wh + (long)c0 * 128 + kc * 32, 128), acc);
      const int col = c0 + (lane & 15);
#pragma unroll
      for (int v = 0; v < 8; ++v) gates[(rb + v) * 512 + col] = acc[v];
    }
    __syncthreads();
    for (int e = tid; e < 2048; e += 256) {
      const int m = e >> 7, j = e & 127;
      const float gi = gates[m * 512 + j];
      const float gf = gates[m * 512 + 128 + j];
      const float gg = gates[m * 512 + 256 + j];
      const float go = gates[m * 512 + 384 + j];
      const float c = sigmoidf_(gf) * cbuf[e] + sigmoidf_(gi) * tanhf(gg);
      cbuf[e] = c;
      hbuf[e] = sigmoidf_(go) * tanhf(c);
    }
    __syncthreads();
  }
  for (int e = tid; e < 2048; e += 256)
    h0[(long)(n0 + (e >> 7)) * 256 + (e & 127)] = hbuf[e];
}

// Backward LSTM single step (h=c=0, so Whh term vanishes): h = sig(o)*tanh(sig(i)*tanh(g))
__global__ __launch_bounds__(256)
void k_bw_elem(const float* __restrict__ gbuf, float* __restrict__ h0) {
  const int i = blockIdx.x * 256 + threadIdx.x;   // exactly N*128
  const int n = i >> 7, j = i & 127;
  const float* g = gbuf + (long)n * 512;
  const float c = sigmoidf_(g[j]) * tanhf(g[256 + j]);
  h0[(long)n * 256 + 128 + j] = sigmoidf_(g[384 + j]) * tanhf(c);
}

// w1 = m0 + (exp(s0_raw)+1e-5)*eps1; per-block KL partials for t=0.
__global__ __launch_bounds__(256)
void k_init_w(const float* __restrict__ p0, const float* __restrict__ eps1,
              float* __restrict__ w1, float* __restrict__ part) {
  __shared__ float sb[768];
  const int tid = threadIdx.x;
  const int i = blockIdx.x * 256 + tid;           // exactly N*DW = 32768
  const int n = i >> 6, j = i & 63;
  const float mean = p0[n * 128 + j];
  const float s = expf(p0[n * 128 + 64 + j]) + 1e-5f;
  w1[i] = mean + s * eps1[i];
  sb[tid] = s; sb[256 + tid] = mean * mean; sb[512 + tid] = logf(s);
  __syncthreads();
  for (int st = 128; st > 0; st >>= 1) {
    if (tid < st) {
      sb[tid] += sb[tid + st];
      sb[256 + tid] += sb[256 + tid + st];
      sb[512 + tid] += sb[512 + tid + st];
    }
    __syncthreads();
  }
  if (tid == 0) {
    part[blockIdx.x * 3 + 0] = sb[0];
    part[blockIdx.x * 3 + 1] = sb[256];
    part[blockIdx.x * 3 + 2] = sb[512];
  }
}

// ------------------------------ main scan ----------------------------------
// Block = 16 batch rows, 16 waves, loops t=1..T-1 with z state in LDS.
__global__ __launch_bounds__(512)
void k_scan(const float* __restrict__ x, const float* __restrict__ u,
            const float* __restrict__ eps,
            const float* __restrict__ wp_w1, const float* __restrict__ wp_b1,
            const float* __restrict__ wp_w2, const float* __restrict__ wp_b2,
            const float* __restrict__ vp_w1, const float* __restrict__ vp_b1,
            const float* __restrict__ vp_w2, const float* __restrict__ vp_b2,
            const float* __restrict__ Am, const float* __restrict__ Bm,
            const float* __restrict__ Cm,
            float* __restrict__ z_all, float* __restrict__ part) {
  extern __shared__ float smem[];
  float* zs    = smem;           // 16*128
  float* h1    = zs + 2048;      // 16*128 (wp hidden, tanh)
  float* hv    = h1 + 2048;      // 16*128 (vp hidden, relu)
  float* pw    = hv + 2048;      // 16*128 (wp output)
  float* va    = pw + 2048;      // 16*16  (vp output, pre-softmax)
  float* alpha = va + 256;       // 16*16
  float* wbuf  = alpha + 256;    // 16*64
  float* Y     = wbuf + 1024;    // [16n][16m][132] padded (33792 floats)
  const int tid = threadIdx.x;
  const int wv = tid >> 5;
  const int n0 = blockIdx.x * 16;
  const int lane = tid & 31;
  const int rb = ((lane >> 4) & 1) << 3;
  for (int e = tid; e < 2048; e += 512)
    zs[e] = z_all[(long)(n0 + (e >> 7)) * T_ * DZ_ + (e & 127)];
  __syncthreads();
  float acc_s = 0.0f, acc_m2 = 0.0f, acc_ls = 0.0f;
  for (int t = 1; t < T_; ++t) {
    const float* xt = x + ((long)n0 * T_ + t) * DX_;
    const float* ut = u + ((long)n0 * T_ + (t - 1)) * DU_;
    const int xld = T_ * DX_, uld = T_ * DU_;
    // phase 1: waves 0-7 -> wp1 (K=416 concat x|z|u), waves 8-15 -> vp1 (K=160 z|u)
    if (wv < 8) {
      const int c0 = wv << 4;
      v8f acc = {};
      for (int kc = 0; kc < 13; ++kc) {
        const int k0 = kc * 32;
        v16h a;
        if (k0 < DX_)            a = load_fragA_f32(xt + k0, xld);
        else if (k0 < DX_ + DZ_) a = load_fragA_f32(zs + (k0 - DX_), 128);
        else                     a = load_fragA_f32(ut + (k0 - DX_ - DZ_), uld);
        acc = wmma_(a, load_fragB_f32(wp_w1 + (long)c0 * 416 + k0, 416), acc);
      }
      const int col = c0 + (lane & 15);
      const float bv = wp_b1[col];
#pragma unroll
      for (int v = 0; v < 8; ++v) h1[(rb + v) * 128 + col] = tanhf(acc[v] + bv);
    } else {
      const int c0 = (wv - 8) << 4;
      v8f acc = {};
      for (int kc = 0; kc < 5; ++kc) {
        const int k0 = kc * 32;
        v16h a = (k0 < DZ_) ? load_fragA_f32(zs + k0, 128)
                            : load_fragA_f32(ut + (k0 - DZ_), uld);
        acc = wmma_(a, load_fragB_f32(vp_w1 + (long)c0 * 160 + k0, 160), acc);
      }
      const int col = c0 + (lane & 15);
      const float bv = vp_b1[col];
#pragma unroll
      for (int v = 0; v < 8; ++v) hv[(rb + v) * 128 + col] = fmaxf(acc[v] + bv, 0.0f);
    }
    __syncthreads();
    // phase 2: wp2 (waves 0-7) and vp2 (wave 8, single 16x16 tile)
    if (wv < 8) {
      const int c0 = wv << 4;
      v8f acc = {};
      for (int k0 = 0; k0 < 128; k0 += 32)
        acc = wmma_(load_fragA_f32(h1 + k0, 128),
                    load_fragB_f32(wp_w2 + (long)c0 * 128 + k0, 128), acc);
      const int col = c0 + (lane & 15);
      const float bv = wp_b2[col];
#pragma unroll
      for (int v = 0; v < 8; ++v) pw[(rb + v) * 128 + col] = acc[v] + bv;
    } else if (wv == 8) {
      v8f acc = {};
      for (int k0 = 0; k0 < 128; k0 += 32)
        acc = wmma_(load_fragA_f32(hv + k0, 128),
                    load_fragB_f32(vp_w2 + k0, 128), acc);
      const int col = lane & 15;
      const float bv = vp_b2[col];
#pragma unroll
      for (int v = 0; v < 8; ++v) va[(rb + v) * 16 + col] = acc[v] + bv;
    }
    __syncthreads();
    // phase 3: w = m + (exp(s)+0.01)*eps, KL partials; softmax -> alpha
    for (int e = tid; e < 1024; e += 512) {
      const int m = e >> 6, j = e & 63;
      const float mean = pw[m * 128 + j];
      const float s = expf(pw[m * 128 + 64 + j]) + 0.01f;
      const float ev = eps[((long)(n0 + m) * T_ + t) * DW_ + j];
      wbuf[e] = mean + s * ev;
      acc_s += s; acc_m2 += mean * mean; acc_ls += logf(s);
    }
    if (tid < 16) {
      float mx = va[tid * 16];
      for (int k = 1; k < 16; ++k) mx = fmaxf(mx, va[tid * 16 + k]);
      float tv[16], sum = 0.0f;
      for (int k = 0; k < 16; ++k) { tv[k] = expf(va[tid * 16 + k] - mx); sum += tv[k]; }
      const float inv = 1.0f / sum;
      for (int k = 0; k < 16; ++k) alpha[tid * 16 + k] = tv[k] * inv;
    }
    __syncthreads();
    // phase 4: stacked Y[n, m*128+i] = [z|u|w] @ [A|B|C]^T  (K=224, 128 tiles)
    v16h af[7];
#pragma unroll
    for (int kc = 0; kc < 7; ++kc) {
      const int k0 = kc * 32;
      if (k0 < 128)      af[kc] = load_fragA_f32(zs + k0, 128);
      else if (k0 < 160) af[kc] = load_fragA_f32(ut + (k0 - 128), uld);
      else               af[kc] = load_fragA_f32(wbuf + (k0 - 160), 64);
    }
    for (int q = 0; q < 8; ++q) {
      const int c0 = ((wv << 3) + q) << 4;
      v8f acc = {};
#pragma unroll
      for (int kc = 0; kc < 7; ++kc)
        acc = wmma_(af[kc], load_fragB_abc(Am, Bm, Cm, c0, kc * 32), acc);
      const int col = c0 + (lane & 15);
      const int mm = col >> 7, ii = col & 127;
#pragma unroll
      for (int v = 0; v < 8; ++v)
        Y[(((rb + v) << 4) + mm) * 132 + ii] = acc[v];
    }
    __syncthreads();
    // phase 5: z_next[n,i] = sum_m alpha[n,m] * Y[n,m,i]
    for (int e = tid; e < 2048; e += 512) {
      const int n = e >> 7, i = e & 127;
      float zv = 0.0f;
#pragma unroll
      for (int mm = 0; mm < 16; ++mm)
        zv += alpha[n * 16 + mm] * Y[((n << 4) + mm) * 132 + i];
      zs[e] = zv;
      z_all[((long)(n0 + n) * T_ + t) * DZ_ + i] = zv;
    }
    __syncthreads();
  }
  // deterministic block reduction of KL partials (reuse Y)
  float* red = Y;
  red[tid] = acc_s; red[512 + tid] = acc_m2; red[1024 + tid] = acc_ls;
  __syncthreads();
  for (int st = 256; st > 0; st >>= 1) {
    if (tid < st) {
      red[tid] += red[tid + st];
      red[512 + tid] += red[512 + tid + st];
      red[1024 + tid] += red[1024 + tid + st];
    }
    __syncthreads();
  }
  if (tid == 0) {
    part[blockIdx.x * 3 + 0] = red[0];
    part[blockIdx.x * 3 + 1] = red[512];
    part[blockIdx.x * 3 + 2] = red[1024];
  }
}

// x_rec tile = r1 @ ob_w2^T + b2; accumulate sum((x - x_rec)^2) per block.
__global__ __launch_bounds__(256)
void k_rec_err(const float* __restrict__ r1, const float* __restrict__ ob_w2,
               const float* __restrict__ ob_b2, const float* __restrict__ x,
               float* __restrict__ part) {
  __shared__ float sb[256];
  const int tid = threadIdx.x;
  const int gw = (int)((blockIdx.x * 256 + tid) >> 5);
  const int nw = (int)((gridDim.x * 256) >> 5);
  const long rows = (long)N_ * T_;
  const long total = (rows >> 4) * (DX_ >> 4);
  const int lane = tid & 31;
  const int rb = ((lane >> 4) & 1) << 3;
  float lsum = 0.0f;
  for (long ti = gw; ti < total; ti += nw) {
    const long r0 = (ti >> 4) << 4;
    const int c0 = (int)(ti & 15) << 4;
    v8f acc = {};
    for (int k0 = 0; k0 < 128; k0 += 32)
      acc = wmma_(load_fragA_f32(r1 + r0 * 128 + k0, 128),
                  load_fragB_f32(ob_w2 + (long)c0 * 128 + k0, 128), acc);
    const int col = c0 + (lane & 15);
    const float bv = ob_b2[col];
#pragma unroll
    for (int v = 0; v < 8; ++v) {
      const float d = x[(r0 + rb + v) * DX_ + col] - (acc[v] + bv);
      lsum += d * d;
    }
  }
  sb[tid] = lsum;
  __syncthreads();
  for (int st = 128; st > 0; st >>= 1) { if (tid < st) sb[tid] += sb[tid + st]; __syncthreads(); }
  if (tid == 0) part[blockIdx.x] = sb[0];
}

// Final fixed-order reduction -> scalar loss.
__global__ void k_final(const float* __restrict__ sq_part, int nsq,
                        const float* __restrict__ scan_part, int nscan,
                        const float* __restrict__ init_part, int ninit,
                        float* __restrict__ out) {
  if (threadIdx.x != 0 || blockIdx.x != 0) return;
  float sq = 0.0f;
  for (int i = 0; i < nsq; ++i) sq += sq_part[i];
  float Ss = 0.0f, Sm2 = 0.0f, Sl = 0.0f;
  for (int i = 0; i < nscan; ++i) {
    Ss += scan_part[3 * i]; Sm2 += scan_part[3 * i + 1]; Sl += scan_part[3 * i + 2];
  }
  for (int i = 0; i < ninit; ++i) {
    Ss += init_part[3 * i]; Sm2 += init_part[3 * i + 1]; Sl += init_part[3 * i + 2];
  }
  const float NTDX = (float)N_ * (float)T_ * (float)DX_;
  const float NTDW = (float)N_ * (float)T_ * (float)DW_;
  const float LOG2PI = 1.8378770664093453f;
  const float kl = 0.5f * (Ss + Sm2 - NTDW - Sl);
  out[0] = 0.5f * sq + 0.5f * NTDX * LOG2PI + kl;   // -(logprob_x - kl)
}

extern "C" void kernel_launch(void* const* d_in, const int* in_sizes, int n_in,
                              void* d_out, int out_size, void* d_ws, size_t ws_size,
                              hipStream_t stream) {
  (void)in_sizes; (void)n_in; (void)out_size; (void)ws_size;
  const float* x      = (const float*)d_in[0];
  const float* u      = (const float*)d_in[1];
  const float* eps1   = (const float*)d_in[2];
  const float* eps    = (const float*)d_in[3];
  const float* fw_Wih = (const float*)d_in[4];
  const float* fw_Whh = (const float*)d_in[5];
  const float* fw_b   = (const float*)d_in[6];
  const float* bw_Wih = (const float*)d_in[7];
  // d_in[8] (bw_Whh) is dead: backward LSTM runs one step from h=0.
  const float* bw_b   = (const float*)d_in[9];
  const float* i1_w = (const float*)d_in[10]; const float* i1_b = (const float*)d_in[11];
  const float* i2_w = (const float*)d_in[12]; const float* i2_b = (const float*)d_in[13];
  const float* z1_w = (const float*)d_in[14]; const float* z1_b = (const float*)d_in[15];
  const float* z2_w = (const float*)d_in[16]; const float* z2_b = (const float*)d_in[17];
  const float* wp_w1 = (const float*)d_in[18]; const float* wp_b1 = (const float*)d_in[19];
  const float* wp_w2 = (const float*)d_in[20]; const float* wp_b2 = (const float*)d_in[21];
  const float* vp_w1 = (const float*)d_in[22]; const float* vp_b1 = (const float*)d_in[23];
  const float* vp_w2 = (const float*)d_in[24]; const float* vp_b2 = (const float*)d_in[25];
  const float* ob_w1 = (const float*)d_in[26]; const float* ob_b1 = (const float*)d_in[27];
  const float* ob_w2 = (const float*)d_in[28]; const float* ob_b2 = (const float*)d_in[29];
  const float* Am = (const float*)d_in[30];
  const float* Bm = (const float*)d_in[31];
  const float* Cm = (const float*)d_in[32];

  float* ws   = (float*)d_ws;
  float* xg   = ws;                                  // [N*T, 512] (reused as r1)
  float* zall = xg + (long)65536 * 512;              // [N, T, 128]
  float* h0   = zall + (long)65536 * 128;            // [N, 256]
  float* gbuf = h0 + 512 * 256;                      // [N, 512]
  float* t1   = gbuf + 512 * 512;                    // [N, 128]
  float* p0   = t1 + 512 * 128;                      // [N, 128]
  float* w1   = p0 + 512 * 128;                      // [N, 64]
  float* t2   = w1 + 512 * 64;                       // [N, 128]
  float* sqp  = t2 + 512 * 128;                      // 1024
  float* scp  = sqp + 1024;                          // 32*3
  float* inp  = scp + 96;                            // 128*3
  float* r1   = xg;                                  // reuse after LSTM consumed xg

  // 1) xg = x @ fw_Wih^T + fw_b   [65536,256]x[256,512]
  k_gemm<<<2048, 256, 0, stream>>>(x, 256, fw_Wih, 256, xg, 512, fw_b, 65536, 512, 256, 0);
  // 2) forward LSTM scan -> h0[:, :128]   (LDS: 49152 + 131072 bytes)
  k_lstm_fw<<<32, 256, 180224, stream>>>(xg, fw_Whh, h0);
  // 3) backward LSTM gates: x[:,T-1] @ bw_Wih^T + bw_b
  k_gemm<<<128, 256, 0, stream>>>(x + (long)(T_ - 1) * 256, (long)T_ * 256,
                                  bw_Wih, 256, gbuf, 512, bw_b, 512, 512, 256, 0);
  // 4) backward LSTM elementwise -> h0[:, 128:]
  k_bw_elem<<<256, 256, 0, stream>>>(gbuf, h0);
  // 5) t1 = relu(h0 @ i1_w^T + i1_b)
  k_gemm<<<32, 256, 0, stream>>>(h0, 256, i1_w, 256, t1, 128, i1_b, 512, 128, 256, 1);
  // 6) p0 = t1 @ i2_w^T + i2_b
  k_gemm<<<32, 256, 0, stream>>>(t1, 128, i2_w, 128, p0, 128, i2_b, 512, 128, 128, 0);
  // 7) w1 = m0 + s0*eps1, KL t=0 partials
  k_init_w<<<128, 256, 0, stream>>>(p0, eps1, w1, inp);
  // 8) t2 = relu(w1 @ z1_w^T + z1_b)
  k_gemm<<<32, 256, 0, stream>>>(w1, 64, z1_w, 64, t2, 128, z1_b, 512, 128, 64, 1);
  // 9) z1 = t2 @ z2_w^T + z2_b  -> z_all[:, 0, :]  (ldc = T*DZ)
  k_gemm<<<32, 256, 0, stream>>>(t2, 128, z2_w, 128, zall, (long)T_ * 128, z2_b, 512, 128, 128, 0);
  // 10) main scan t=1..T-1   (LDS: 43520 floats = 174080 bytes)
  k_scan<<<32, 512, 174080, stream>>>(x, u, eps, wp_w1, wp_b1, wp_w2, wp_b2,
                                      vp_w1, vp_b1, vp_w2, vp_b2, Am, Bm, Cm,
                                      zall, scp);
  // 11) r1 = relu(z_all @ ob_w1^T + ob_b1)
  k_gemm<<<1024, 256, 0, stream>>>(zall, 128, ob_w1, 128, r1, 128, ob_b1, 65536, 128, 128, 1);
  // 12) fused x_rec + squared error partials
  k_rec_err<<<1024, 256, 0, stream>>>(r1, ob_w2, ob_b2, x, sqp);
  // 13) final scalar loss
  k_final<<<1, 1, 0, stream>>>(sqp, 1024, scp, 32, inp, 128, (float*)d_out);
}
